// Transformer_48644799594659
// MI455X (gfx1250) — compile-verified, matrix operand order checked by reading, not run on hardware
//
#include <hip/hip_runtime.h>

#define BATCH 4
#define SEQ   1024
#define HID   1024
#define NH    16
#define HD    64
#define NBUCK 63
#define SCALE 0.125f

typedef __attribute__((ext_vector_type(16))) __bf16 v16bf;
typedef __attribute__((ext_vector_type(8)))  __bf16 v8bf;
typedef __attribute__((ext_vector_type(8)))  float  v8f;
typedef int v4i __attribute__((vector_size(16)));

union F8  { v8f  v; float  f[8]; };
union BF8 { v8bf v; __bf16 e[8]; };

// --- CDNA5 async global->LDS staging (guarded: falls back to sync copy) -----
#if defined(__AMDGCN__) && __has_builtin(__builtin_amdgcn_global_load_async_to_lds_b128) && __has_builtin(__builtin_amdgcn_s_wait_asynccnt)
#define USE_ASYNC_LDS 1
#else
#define USE_ASYNC_LDS 0
#endif

__device__ __forceinline__ void stage16(const void* gsrc, void* ldst) {
#if USE_ASYNC_LDS
  __builtin_amdgcn_global_load_async_to_lds_b128(
      (__attribute__((address_space(1))) v4i*)gsrc,
      (__attribute__((address_space(3))) v4i*)ldst, 0, 0);
#else
  *(v8bf*)ldst = *(const v8bf*)gsrc;
#endif
}

__device__ __forceinline__ void stage_fence() {
#if USE_ASYNC_LDS
  __builtin_amdgcn_s_wait_asynccnt(0);
#endif
}

// ---------------------------------------------------------------------------
// WMMA 16-bit operand fragment layout (ISA 7.12.2), 16x32 tile (512 elems):
//   lanes 0-15 hold K {0..7, 16..23}, lanes 16-31 hold K {8..15, 24..31};
//   per-lane element i -> K = (i/8)*16 + (lane>=16 ? 8 : 0) + (i%8).
// Tiles are stored in LDS *pre-swizzled* so each lane's 16 values are
// contiguous: one 32B read per lane (2x ds_load_b128), no packing moves.
// 8 consecutive k with the same (k&8) map to 8 contiguous elements -> 16B
// vector (or async b128) staging stores.
// ---------------------------------------------------------------------------
__device__ __forceinline__ int swz_idx(int mn, int k) {
  int rem = k & 15;
  int hi  = (rem & 8) ? 16 : 0;
  int i   = ((k >> 4) << 3) + (rem & 7);
  return (mn + hi) * 16 + i;
}

__device__ __forceinline__ v16bf load_frag(const __bf16* region, int lane) {
  return *(const v16bf*)(region + lane * 16);
}

__device__ __forceinline__ v8f wmma_bf16(v16bf a, v16bf b, v8f c) {
  return __builtin_amdgcn_wmma_f32_16x16x32_bf16(false, a, false, b, (short)0, c, false, false);
}

// ---------------------------------------------------------------------------
// Generic tiled GEMM: C[M,N] = A[M,K] @ B[K,N] + bias[N]
// Tile 128x64, K-step 32, 8 waves (each wave: 32x32 = 2x2 WMMA subtiles)
// ---------------------------------------------------------------------------
template<bool OUT_BF16>
__global__ void __launch_bounds__(256) gemm_bias_wmma(
    const float* __restrict__ A, const float* __restrict__ B,
    const float* __restrict__ bias, void* __restrict__ Cv,
    int M, int N, int K)
{
  __shared__ alignas(32) __bf16 Asw[8 * 512];   // 8 row-subtiles x (16x32) frag
  __shared__ alignas(32) __bf16 Bsw[4 * 512];   // 4 col-subtiles x (16x32) frag
  const int tn = blockIdx.x * 64;
  const int tm = blockIdx.y * 128;
  const int tid = threadIdx.x;
  const int lane = tid & 31, wave = tid >> 5;
  const int wm = (wave & 3) * 32, wn = (wave >> 2) * 32;
  const int subm = wm >> 4, subn = wn >> 4;

  F8 acc[2][2] = {};

  for (int k0 = 0; k0 < K; k0 += 32) {
    __syncthreads();
    // A tile 128x32: contiguous axis == K -> vector load + vector swizzled store
#pragma unroll
    for (int g = tid; g < (128 * 32) / 8; g += 256) {
      int idx = g * 8;
      int r = idx >> 5, c = idx & 31;
      int gr = tm + r;
      BF8 u;
#pragma unroll
      for (int e = 0; e < 8; ++e)
        u.e[e] = (__bf16)((gr < M) ? A[(size_t)gr * K + k0 + c + e] : 0.0f);
      *(v8bf*)(&Asw[(r >> 4) * 512 + swz_idx(r & 15, c)]) = u.v;
    }
    // B tile 32x64: contiguous axis == N (lane axis) -> scalar swizzled stores
#pragma unroll
    for (int g = tid; g < (32 * 64) / 8; g += 256) {
      int idx = g * 8;
      int k = idx >> 6, n0 = idx & 63;
#pragma unroll
      for (int e = 0; e < 8; ++e) {
        int n = n0 + e;
        float v = B[(size_t)(k0 + k) * N + tn + n];
        Bsw[(n >> 4) * 512 + swz_idx(n & 15, k)] = (__bf16)v;
      }
    }
    if (k0 + 32 < K) {  // hint next A tile (global_prefetch_b8)
      int gr = tm + (tid >> 1);
      if (gr < M) __builtin_prefetch(&A[(size_t)gr * K + k0 + 32 + (tid & 1) * 16], 0, 1);
    }
    __syncthreads();

    v16bf af0 = load_frag(&Asw[(subm + 0) * 512], lane);
    v16bf af1 = load_frag(&Asw[(subm + 1) * 512], lane);
    v16bf bf0 = load_frag(&Bsw[(subn + 0) * 512], lane);
    v16bf bf1 = load_frag(&Bsw[(subn + 1) * 512], lane);
    acc[0][0].v = wmma_bf16(af0, bf0, acc[0][0].v);
    acc[0][1].v = wmma_bf16(af0, bf1, acc[0][1].v);
    acc[1][0].v = wmma_bf16(af1, bf0, acc[1][0].v);
    acc[1][1].v = wmma_bf16(af1, bf1, acc[1][1].v);
  }

  const int rowoff = (lane & 16) ? 8 : 0;
  const int coll = lane & 15;
#pragma unroll
  for (int i = 0; i < 2; ++i)
#pragma unroll
    for (int j = 0; j < 2; ++j) {
      int col = tn + wn + j * 16 + coll;
      float bb = bias[col];
#pragma unroll
      for (int r = 0; r < 8; ++r) {
        int grow = tm + wm + i * 16 + rowoff + r;
        if (grow < M) {
          float v = acc[i][j].f[r] + bb;
          if (OUT_BF16) ((__bf16*)Cv)[(size_t)grow * N + col] = (__bf16)v;
          else          ((float*)Cv)[(size_t)grow * N + col] = v;
        }
      }
    }
}

// ---------------------------------------------------------------------------
// Per-head relative-table GEMM: C[bh,m,r] = SCALE * sum_d A[b,m,h*64+d]*T[r,h*64+d]
// A: bf16 [B*S, HID]; T: bf16 [NBUCK, HID]; C: f32 [B*H, S, 64] (col 63 pad)
// ---------------------------------------------------------------------------
__global__ void __launch_bounds__(256) relproj_wmma(
    const __bf16* __restrict__ A, const __bf16* __restrict__ T,
    float* __restrict__ C)
{
  __shared__ alignas(32) __bf16 Asw[16 * 512];  // (8 row-sub x 2 kstep) frags
  __shared__ alignas(32) __bf16 Tsw[8 * 512];   // (4 col-sub x 2 kstep) frags
  const int bh = blockIdx.y;
  const int b = bh >> 4, h = bh & 15;
  const int tm = blockIdx.x * 128;
  const int tid = threadIdx.x;
  const int lane = tid & 31, wave = tid >> 5;
  const int wm = (wave & 3) * 32, wn = (wave >> 2) * 32;
  const int subm = wm >> 4, subn = wn >> 4;

  // A tile 128x64 (contig axis == d == K): async b128 global->LDS staging
#pragma unroll
  for (int g = tid; g < (128 * 64) / 8; g += 256) {
    int idx = g * 8;
    int r = idx >> 6, c = idx & 63;
    stage16(&A[((size_t)b * SEQ + tm + r) * HID + h * HD + c],
            &Asw[((r >> 4) * 2 + (c >> 5)) * 512 + swz_idx(r & 15, c & 31)]);
  }
  // T tile 64x64 (rows rr are lane axis, contig axis d == K)
#pragma unroll
  for (int g = tid; g < (64 * 64) / 8; g += 256) {
    int idx = g * 8;
    int rr = idx >> 6, d0 = idx & 63;
    __bf16* dst = &Tsw[((rr >> 4) * 2 + (d0 >> 5)) * 512 + swz_idx(rr & 15, d0 & 31)];
    if (rr < NBUCK) {
      stage16(&T[(size_t)rr * HID + h * HD + d0], dst);
    } else {
      BF8 u;
#pragma unroll
      for (int e = 0; e < 8; ++e) u.e[e] = (__bf16)0.0f;
      *(v8bf*)dst = u.v;
    }
  }
  stage_fence();
  __syncthreads();

  F8 acc[2][2] = {};
#pragma unroll
  for (int s = 0; s < 2; ++s) {
    v16bf af0 = load_frag(&Asw[((subm + 0) * 2 + s) * 512], lane);
    v16bf af1 = load_frag(&Asw[((subm + 1) * 2 + s) * 512], lane);
    v16bf bf0 = load_frag(&Tsw[((subn + 0) * 2 + s) * 512], lane);
    v16bf bf1 = load_frag(&Tsw[((subn + 1) * 2 + s) * 512], lane);
    acc[0][0].v = wmma_bf16(af0, bf0, acc[0][0].v);
    acc[0][1].v = wmma_bf16(af0, bf1, acc[0][1].v);
    acc[1][0].v = wmma_bf16(af1, bf0, acc[1][0].v);
    acc[1][1].v = wmma_bf16(af1, bf1, acc[1][1].v);
  }

  const int rowoff = (lane & 16) ? 8 : 0;
  const int coll = lane & 15;
#pragma unroll
  for (int i = 0; i < 2; ++i)
#pragma unroll
    for (int j = 0; j < 2; ++j)
#pragma unroll
      for (int r = 0; r < 8; ++r) {
        int m = tm + wm + i * 16 + rowoff + r;
        C[((size_t)bh * SEQ + m) * 64 + wn + j * 16 + coll] = SCALE * acc[i][j].f[r];
      }
}

// ---------------------------------------------------------------------------
// Flash attention with relative-position gathers.
// Grid: (SEQ/128, B*H). Each of 8 waves owns 16 query rows; K/V streamed in
// 64-wide chunks through swizzled LDS. q staged (swizzled, async) through the
// P region once; P written by each wave directly in A-fragment layout.
// ---------------------------------------------------------------------------
__global__ void __launch_bounds__(256) attention_wmma(
    const __bf16* __restrict__ qp, const __bf16* __restrict__ kp,
    const __bf16* __restrict__ vp,
    const float* __restrict__ c2p, const float* __restrict__ p2c,
    const int* __restrict__ qpi, const int* __restrict__ kpi,
    float* __restrict__ outp)
{
  __shared__ alignas(32) __bf16 Ksw[8 * 512];    // (4 n-sub x 2 kstep) B frags of K^T
  __shared__ alignas(32) __bf16 Vsw[8 * 512];    // (4 n-sub x 2 kstep) B frags of V
  __shared__ alignas(32) __bf16 Psw[8 * 1024];   // per-wave A frags (q first, then P)

  const int bh = blockIdx.y;
  const int b = bh >> 4, h = bh & 15;
  const int q0 = blockIdx.x * 128;
  const int tid = threadIdx.x;
  const int lane = tid & 31, wave = tid >> 5;
  const int rowoff = (lane & 16) ? 8 : 0;
  const int coll = lane & 15;

  // stage q tile 128x64 swizzled into Psw (contig axis == d == K): async b128
#pragma unroll
  for (int g = tid; g < (128 * 64) / 8; g += 256) {
    int idx = g * 8;
    int r = idx >> 6, c = idx & 63;
    stage16(&qp[((size_t)b * SEQ + q0 + r) * HID + h * HD + c],
            &Psw[(r >> 4) * 1024 + (c >> 5) * 512 + swz_idx(r & 15, c & 31)]);
  }
  stage_fence();
  __syncthreads();
  v16bf aq0 = load_frag(&Psw[wave * 1024 + 0], lane);
  v16bf aq1 = load_frag(&Psw[wave * 1024 + 512], lane);

  float mi[8], li[8];
  F8 Oacc[4] = {};
#pragma unroll
  for (int r = 0; r < 8; ++r) { mi[r] = -3.0e38f; li[r] = 0.0f; }

  for (int k0 = 0; k0 < SEQ; k0 += 64) {
    __syncthreads();
    // K tile: B-operand of q@k^T has n = k-row (lane axis), K = d (contig)
    //         -> async b128 swizzled staging
    // V tile: B-operand of P@V has n = d (contig axis), K = k-row
    //         -> vector global load + scalar swizzled scatter
#pragma unroll
    for (int g = tid; g < (64 * 64) / 8; g += 256) {
      int idx = g * 8;
      int r = idx >> 6, c = idx & 63;              // r = k row, c = d base
      size_t gbase = ((size_t)b * SEQ + k0 + r) * HID + h * HD + c;
      stage16(&kp[gbase],
              &Ksw[((r >> 4) * 2 + (c >> 5)) * 512 + swz_idx(r & 15, c & 31)]);
      BF8 vv; vv.v = *(const v8bf*)(&vp[gbase]);
#pragma unroll
      for (int e = 0; e < 8; ++e) {
        int d = c + e;
        Vsw[((d >> 4) * 2 + (r >> 5)) * 512 + swz_idx(d & 15, r & 31)] = vv.e[e];
      }
    }
    stage_fence();
    __syncthreads();

    // S = q @ k^T
    F8 S[4];
#pragma unroll
    for (int kk = 0; kk < 4; ++kk) {
      v16bf bk0 = load_frag(&Ksw[(kk * 2 + 0) * 512], lane);
      v16bf bk1 = load_frag(&Ksw[(kk * 2 + 1) * 512], lane);
      v8f s = {};
      s = wmma_bf16(aq0, bk0, s);
      s = wmma_bf16(aq1, bk1, s);
      S[kk].v = s;
    }

    // scale + relative-position gathers + online softmax
    float newmax[8];
#pragma unroll
    for (int r = 0; r < 8; ++r) {
      const int qg = q0 + wave * 16 + rowoff + r;
      float rmax = -3.0e38f;
#pragma unroll
      for (int kk = 0; kk < 4; ++kk) {
        const int kg = k0 + kk * 16 + coll;
        const size_t ij = (size_t)qg * SEQ + kg;
        float s = S[kk].f[r] * SCALE;
        s += c2p[((size_t)bh * SEQ + qg) * 64 + kpi[ij]];
        s += p2c[((size_t)bh * SEQ + kg) * 64 + qpi[ij]];
        S[kk].f[r] = s;
        rmax = fmaxf(rmax, s);
      }
#pragma unroll
      for (int off = 8; off > 0; off >>= 1)
        rmax = fmaxf(rmax, __shfl_xor(rmax, off, 16));
      newmax[r] = fmaxf(mi[r], rmax);
    }

#pragma unroll
    for (int r = 0; r < 8; ++r) {
      const float alpha = __expf(mi[r] - newmax[r]);
      float rsum = 0.0f;
#pragma unroll
      for (int kk = 0; kk < 4; ++kk) {
        float p = __expf(S[kk].f[r] - newmax[r]);
        S[kk].f[r] = p;
        rsum += p;
      }
#pragma unroll
      for (int off = 8; off > 0; off >>= 1)
        rsum += __shfl_xor(rsum, off, 16);
      li[r] = li[r] * alpha + rsum;
      mi[r] = newmax[r];
#pragma unroll
      for (int dd = 0; dd < 4; ++dd) Oacc[dd].f[r] *= alpha;
    }

    // write P directly in swizzled A-fragment layout (wave-local, in-order LDS)
#pragma unroll
    for (int r = 0; r < 8; ++r) {
      const int m = rowoff + r;
#pragma unroll
      for (int kk = 0; kk < 4; ++kk) {
        int k32 = ((kk & 1) << 4) | coll;
        Psw[wave * 1024 + (kk >> 1) * 512 + swz_idx(m, k32)] = (__bf16)S[kk].f[r];
      }
    }
    v16bf pf0 = load_frag(&Psw[wave * 1024 + 0], lane);
    v16bf pf1 = load_frag(&Psw[wave * 1024 + 512], lane);
#pragma unroll
    for (int dd = 0; dd < 4; ++dd) {
      v16bf vb0 = load_frag(&Vsw[(dd * 2 + 0) * 512], lane);
      v16bf vb1 = load_frag(&Vsw[(dd * 2 + 1) * 512], lane);
      Oacc[dd].v = wmma_bf16(pf0, vb0, Oacc[dd].v);
      Oacc[dd].v = wmma_bf16(pf1, vb1, Oacc[dd].v);
    }
  }

  // normalize and write attention output (f32, [B,S,HID] head slice)
#pragma unroll
  for (int dd = 0; dd < 4; ++dd)
#pragma unroll
    for (int r = 0; r < 8; ++r) {
      const int qg = q0 + wave * 16 + rowoff + r;
      outp[((size_t)b * SEQ + qg) * HID + h * HD + dd * 16 + coll] =
          Oacc[dd].f[r] / li[r];
    }
}

// ---------------------------------------------------------------------------
extern "C" void kernel_launch(void* const* d_in, const int* in_sizes, int n_in,
                              void* d_out, int out_size, void* d_ws, size_t ws_size,
                              hipStream_t stream) {
  (void)in_sizes; (void)n_in; (void)out_size; (void)ws_size;
  const float* queries = (const float*)d_in[0];
  const float* keys    = (const float*)d_in[1];
  const float* values  = (const float*)d_in[2];
  const float* qrel    = (const float*)d_in[3];
  const float* krel    = (const float*)d_in[4];
  const float* Wq = (const float*)d_in[5];
  const float* bq = (const float*)d_in[6];
  const float* Wk = (const float*)d_in[7];
  const float* bk = (const float*)d_in[8];
  const float* Wv = (const float*)d_in[9];
  const float* bv = (const float*)d_in[10];
  const float* Wo = (const float*)d_in[11];
  const float* bo = (const float*)d_in[12];
  const int* qpi = (const int*)d_in[13];
  const int* kpi = (const int*)d_in[14];
  float* out = (float*)d_out;

  char* ws = (char*)d_ws;
  size_t off = 0;
  auto take = [&](size_t bytes) -> void* {
    void* p = ws + off;
    off += (bytes + 255) & ~(size_t)255;
    return p;
  };
  const size_t MROWS = (size_t)BATCH * SEQ;           // 4096
  __bf16* qpb  = (__bf16*)take(MROWS * HID * 2);
  __bf16* kpb  = (__bf16*)take(MROWS * HID * 2);
  __bf16* vpb  = (__bf16*)take(MROWS * HID * 2);
  __bf16* rqb  = (__bf16*)take((size_t)64 * HID * 2);
  __bf16* rkb  = (__bf16*)take((size_t)64 * HID * 2);
  float*  c2p  = (float*)take((size_t)BATCH * NH * SEQ * 64 * 4);
  float*  p2c  = (float*)take((size_t)BATCH * NH * SEQ * 64 * 4);
  float*  attn = (float*)take(MROWS * HID * 4);

  dim3 blk(256);
  dim3 gBig(HID / 64, (unsigned)((MROWS + 127) / 128));
  gemm_bias_wmma<true><<<gBig, blk, 0, stream>>>(queries, Wq, bq, qpb, (int)MROWS, HID, HID);
  gemm_bias_wmma<true><<<gBig, blk, 0, stream>>>(keys,    Wk, bk, kpb, (int)MROWS, HID, HID);
  gemm_bias_wmma<true><<<gBig, blk, 0, stream>>>(values,  Wv, bv, vpb, (int)MROWS, HID, HID);

  dim3 gRel(HID / 64, 1);
  gemm_bias_wmma<true><<<gRel, blk, 0, stream>>>(qrel, Wq, bq, rqb, NBUCK, HID, HID);
  gemm_bias_wmma<true><<<gRel, blk, 0, stream>>>(krel, Wk, bk, rkb, NBUCK, HID, HID);

  dim3 gHead(SEQ / 128, BATCH * NH);
  relproj_wmma<<<gHead, blk, 0, stream>>>(qpb, rkb, c2p);   // c2p = SCALE * q @ rel_k^T
  relproj_wmma<<<gHead, blk, 0, stream>>>(kpb, rqb, p2c);   // p2c = SCALE * k @ rel_q^T

  attention_wmma<<<gHead, blk, 0, stream>>>(qpb, kpb, vpb, c2p, p2c, qpi, kpi, attn);

  gemm_bias_wmma<false><<<gBig, blk, 0, stream>>>(attn, Wo, bo, out, (int)MROWS, HID, HID);
}
